// SelectiveWKV_1B_70806830842348
// MI455X (gfx1250) — matrix operand order repaired and checked by reading, not compiled, not run on hardware
//
#include <hip/hip_runtime.h>
#include <cstddef>
#include <cstdint>

// Problem constants (match reference)
#define BATCH   8
#define SEQ     2048
#define DM      2048      // d_model
#define HEADS   32
#define HSZ     64

typedef _Float16 v8h   __attribute__((ext_vector_type(8)));
typedef _Float16 v16h  __attribute__((ext_vector_type(16)));
typedef float    v8f   __attribute__((ext_vector_type(8)));

// gcc-style vectors matching the builtin parameter types reported by hipcc
typedef int          gi32x4 __attribute__((vector_size(16)));
typedef int          gi32x8 __attribute__((vector_size(32)));
typedef unsigned int gu32x4 __attribute__((vector_size(16)));

enum { EPI_F16 = 0, EPI_F16_SIG = 1, EPI_F16_BIAS_SIG = 2, EPI_F32 = 3 };

// ---- CDNA5 async / TDM feature probes (compile-safe fallbacks) -------------
#if defined(__gfx1250__) && __has_builtin(__builtin_amdgcn_global_load_async_to_lds_b128) && __has_builtin(__builtin_amdgcn_s_wait_asynccnt)
#define HAVE_ASYNC_LDS 1
#else
#define HAVE_ASYNC_LDS 0
#endif

#if HAVE_ASYNC_LDS && __has_builtin(__builtin_amdgcn_tensor_load_to_lds) && __has_builtin(__builtin_amdgcn_s_wait_tensorcnt)
#define HAVE_TDM 1
#else
#define HAVE_TDM 0
#endif

#if HAVE_TDM
#define AS_PER_STAGE 2   // async b128 ops per thread per stage (A tile only)
#else
#define AS_PER_STAGE 3   // A tile (2) + B tile (1)
#endif

__device__ __forceinline__ float sigmoid_f(float x) {
    return 1.0f / (1.0f + __expf(-x));
}

// ---------------------------------------------------------------------------
// Transpose fp32 W[C][C] -> f16 Wt[N][K] (Wt[n][k] = W[k][n]) via LDS tile.
// ---------------------------------------------------------------------------
__global__ __launch_bounds__(256)
void transpose_to_f16(const float* __restrict__ W, _Float16* __restrict__ Wt, int C)
{
    __shared__ float tile[32][33];
    const int bx = blockIdx.x * 32;      // n tile
    const int by = blockIdx.y * 32;      // k tile
    const int tx = threadIdx.x & 31;
    const int ty = threadIdx.x >> 5;     // 0..7
    #pragma unroll
    for (int i = ty; i < 32; i += 8)
        tile[i][tx] = W[(size_t)(by + i) * C + (bx + tx)];
    __syncthreads();
    #pragma unroll
    for (int i = ty; i < 32; i += 8)
        Wt[(size_t)(bx + i) * C + (by + tx)] = (_Float16)tile[tx][i];
}

// ---------------------------------------------------------------------------
// LayerNorm over last dim (C), write f16. One block per (b,t) row.
// ---------------------------------------------------------------------------
__global__ __launch_bounds__(256)
void layernorm_to_f16(const float* __restrict__ x,
                      const float* __restrict__ gamma,
                      const float* __restrict__ beta,
                      _Float16* __restrict__ xn, int C)
{
    const int row = blockIdx.x;
    const float* xr = x + (size_t)row * C;
    __shared__ float s1[256], s2[256];
    float sum = 0.f, sq = 0.f;
    for (int c = threadIdx.x; c < C; c += 256) {
        float v = xr[c];
        sum += v; sq += v * v;
    }
    s1[threadIdx.x] = sum; s2[threadIdx.x] = sq;
    __syncthreads();
    for (int off = 128; off > 0; off >>= 1) {
        if (threadIdx.x < off) {
            s1[threadIdx.x] += s1[threadIdx.x + off];
            s2[threadIdx.x] += s2[threadIdx.x + off];
        }
        __syncthreads();
    }
    const float inv = 1.0f / (float)C;
    const float mu  = s1[0] * inv;
    const float var = s2[0] * inv - mu * mu;
    const float rstd = rsqrtf(var + 1e-5f);
    for (int c = threadIdx.x; c < C; c += 256) {
        float v = (xr[c] - mu) * rstd * gamma[c] + beta[c];
        xn[(size_t)row * C + c] = (_Float16)v;
    }
}

// ---------------------------------------------------------------------------
// WMMA GEMM, LDS-staged + double-buffered.
//   D = A[M,K=2048](f16 row-major) x Bt[N,K](f16, weights pre-transposed)
// Block = 256 threads = 8 waves; block tile 128(M) x 64(N); K-step 32.
// A tile -> LDS via GLOBAL_LOAD_ASYNC_TO_LDS_B128 (ASYNCcnt)
// B tile -> LDS via TENSOR_LOAD_TO_LDS (TDM, TENSORcnt), wave 0 issues.
// WMMAs consume fragments from LDS (ds_load); fp32 accumulate.
// ---------------------------------------------------------------------------
template<int EPI>
__global__ __launch_bounds__(256)
void gemm_wmma_f16(const _Float16* __restrict__ A,
                   const _Float16* __restrict__ Bt,
                   const float*    __restrict__ bias,   // [N] (only BIAS_SIG)
                   _Float16*       __restrict__ outh,   // f16 out
                   float*          __restrict__ outf,   // f32 out
                   int M, int N)
{
    (void)M;
    constexpr int K  = 2048;
    constexpr int S  = K / 32;      // pipeline stages
    constexpr int AP = 40;          // padded row length in halfs (32 + 8)
#if HAVE_ASYNC_LDS
    constexpr int NBUF = 2;
#else
    constexpr int NBUF = 1;
#endif
    __shared__ __align__(16) _Float16 As[NBUF][128][AP];
    __shared__ __align__(16) _Float16 Bs[NBUF][64][AP];

    const int tid  = threadIdx.x;
    const int lane = tid & 31;
    const int wave = tid >> 5;
    const int n0   = blockIdx.x * 64;
    const int mBase = blockIdx.y * 128;
    const int l15  = lane & 15;
    const int hi   = lane >> 4;        // 0 or 1
    const int kbA  = hi * 16;          // A sub-K offset (halfs), contiguous like B
    const int kgB  = hi * 16;          // B sub-K offset (halfs), contiguous like A

    v8f acc0 = {}, acc1 = {}, acc2 = {}, acc3 = {};

    auto compute = [&](int buf) {
        const _Float16* arow = &As[buf][wave * 16 + l15][0];
        v8h alo = *(const v8h*)(arow + kbA);
        v8h ahi = *(const v8h*)(arow + kbA + 8);
        v16h a = __builtin_shufflevector(alo, ahi,
                                         0,1,2,3,4,5,6,7,8,9,10,11,12,13,14,15);
        v16h b0 = *(const v16h*)&Bs[buf][ 0 + l15][kgB];
        v16h b1 = *(const v16h*)&Bs[buf][16 + l15][kgB];
        v16h b2 = *(const v16h*)&Bs[buf][32 + l15][kgB];
        v16h b3 = *(const v16h*)&Bs[buf][48 + l15][kgB];
        acc0 = __builtin_amdgcn_wmma_f32_16x16x32_f16(false, a, false, b0, (short)0, acc0, false, false);
        acc1 = __builtin_amdgcn_wmma_f32_16x16x32_f16(false, a, false, b1, (short)0, acc1, false, false);
        acc2 = __builtin_amdgcn_wmma_f32_16x16x32_f16(false, a, false, b2, (short)0, acc2, false, false);
        acc3 = __builtin_amdgcn_wmma_f32_16x16x32_f16(false, a, false, b3, (short)0, acc3, false, false);
    };

#if HAVE_ASYNC_LDS
    auto async_b128 = [](const _Float16* gsrc, _Float16* ldst) {
        __builtin_amdgcn_global_load_async_to_lds_b128(
            (__attribute__((address_space(1))) gi32x4*)const_cast<_Float16*>(gsrc),
            (__attribute__((address_space(3))) gi32x4*)ldst,
            0, 0);
    };

    auto issue_stage = [&](int s, int buf) {
        const int k0 = s * 32;
        // A tile: 128 rows x 64B = 512 x 16B chunks, 2 per thread (ASYNCcnt)
        #pragma unroll
        for (int c = tid; c < 512; c += 256) {
            const int row = c >> 2;
            const int ko  = (c & 3) * 8;
            async_b128(A + (size_t)(mBase + row) * K + k0 + ko, &As[buf][row][ko]);
        }
#if HAVE_TDM
        // B tile via Tensor Data Mover: 64x32 f16 tile of Bt, padded rows.
        if (wave == 0) {
            const unsigned ldsb =
                (unsigned)(uintptr_t)(__attribute__((address_space(3))) void*)&Bs[buf][0][0];
            const unsigned long long ga =
                (unsigned long long)(uintptr_t)(Bt + (size_t)n0 * K + k0);
            gu32x4 g0;
            g0[0] = 1u;                                   // count=1 (valid user D#)
            g0[1] = ldsb;                                 // lds_addr
            g0[2] = (unsigned)ga;                         // global_addr[31:0]
            g0[3] = (unsigned)((ga >> 32) & 0x01FFFFFFu)  // global_addr[56:32]
                  | (2u << 30);                           // type = 2 (image)
            gi32x8 g1;
            g1[0] = (int)((1u << 16)      // data_size = 2 bytes
                        | (1u << 20)      // pad_enable
                        | (3u << 22)      // pad_interval: every 16 DWORDs (64B row)
                        | (3u << 25));    // pad_amount: 4 DWORDs (8 halfs)
            g1[1] = (int)(((unsigned)K & 0xFFFFu) << 16);             // tensor_dim0 lo
            g1[2] = (int)(((unsigned)K >> 16) | (((unsigned)N & 0xFFFFu) << 16)); // dim0 hi | dim1 lo
            g1[3] = (int)(((unsigned)N >> 16) | (32u << 16));         // dim1 hi | tile_dim0=32
            g1[4] = (int)(64u);                                       // tile_dim1=64, tile_dim2=0
            const unsigned long long d1s = (unsigned long long)K * (unsigned long long)N;
            g1[5] = (int)(unsigned)K;                                 // tensor_dim0_stride lo32
            g1[6] = (int)(((unsigned)(d1s & 0xFFFFull)) << 16);       // stride0 hi=0 | stride1 lo16
            g1[7] = (int)(unsigned)(d1s >> 16);                       // stride1 [47:16]
            gi32x4 z4 = {0, 0, 0, 0};
            gi32x8 z8 = {0, 0, 0, 0, 0, 0, 0, 0};
            __builtin_amdgcn_tensor_load_to_lds(g0, g1, z4, z4, z8, 0);
        }
#else
        // B tile: 64 rows x 64B = 256 x 16B chunks, 1 per thread (ASYNCcnt)
        {
            const int row = tid >> 2;
            const int ko  = (tid & 3) * 8;
            async_b128(Bt + (size_t)(n0 + row) * K + k0 + ko, &Bs[buf][row][ko]);
        }
#endif
    };

    issue_stage(0, 0);
    for (int s = 0; s < S; ++s) {
        const int buf = s & 1;
        if (s + 1 < S) {
            issue_stage(s + 1, buf ^ 1);
            __builtin_amdgcn_s_wait_asynccnt(AS_PER_STAGE); // stage s A-copies done
#if HAVE_TDM
            if (wave == 0) __builtin_amdgcn_s_wait_tensorcnt(1); // stage s B-tile done
#endif
        } else {
            __builtin_amdgcn_s_wait_asynccnt(0);
#if HAVE_TDM
            if (wave == 0) __builtin_amdgcn_s_wait_tensorcnt(0);
#endif
        }
        __syncthreads();      // stage-s LDS visible to all waves
        compute(buf);
        __syncthreads();      // all readers done before buf is overwritten
    }
#else
    // Fallback: synchronous global->LDS staging, single buffer.
    for (int s = 0; s < S; ++s) {
        const int k0 = s * 32;
        __syncthreads();
        #pragma unroll
        for (int c = tid; c < 512; c += 256) {
            const int row = c >> 2;
            const int ko  = (c & 3) * 8;
            *(v8h*)&As[0][row][ko] = *(const v8h*)(A + (size_t)(mBase + row) * K + k0 + ko);
        }
        {
            const int row = tid >> 2;
            const int ko  = (tid & 3) * 8;
            *(v8h*)&Bs[0][row][ko] = *(const v8h*)(Bt + (size_t)(n0 + row) * K + k0 + ko);
        }
        __syncthreads();
        compute(0);
    }
#endif

    // Epilogue. D layout: VGPR i -> row = mBase + wave*16 + i + 8*hi, col = n0 + j*16 + l15
    v8f accs[4] = { acc0, acc1, acc2, acc3 };
    const int m0 = mBase + wave * 16;
    #pragma unroll
    for (int j = 0; j < 4; ++j) {
        const int col = n0 + j * 16 + l15;
        float bval = 0.f;
        if (EPI == EPI_F16_BIAS_SIG) bval = bias[col];
        #pragma unroll
        for (int i = 0; i < 8; ++i) {
            const int r = m0 + i + 8 * hi;
            const size_t off = (size_t)r * N + col;
            float v = accs[j][i];
            if (EPI == EPI_F32) {
                outf[off] = v;
            } else {
                if (EPI == EPI_F16_BIAS_SIG) v = sigmoid_f(v + bval);
                else if (EPI == EPI_F16_SIG) v = sigmoid_f(v);
                outh[off] = (_Float16)v;
            }
        }
    }
}

// ---------------------------------------------------------------------------
// Selective WKV scan. One block per (b,h). State 64x64 fp32 in registers.
// ---------------------------------------------------------------------------
__global__ __launch_bounds__(256)
void wkv_scan(const _Float16* __restrict__ wbuf,
              const _Float16* __restrict__ kbuf,
              const _Float16* __restrict__ vbuf,
              const _Float16* __restrict__ rbuf,
              _Float16* __restrict__ ybuf,
              float* __restrict__ state_out,
              int T, int C, int H)
{
    const int bh  = blockIdx.x;
    const int b   = bh / H;
    const int h   = bh % H;
    const int tid = threadIdx.x;
    const int col = tid & 63;
    const int rg  = tid >> 6;

    __shared__ float sw[64], sk[64], sv[64], sr[64];
    __shared__ float sred[256];

    float st[16];
    #pragma unroll
    for (int i = 0; i < 16; ++i) st[i] = 0.f;

    const size_t base = (size_t)b * T * C + (size_t)h * HSZ;

    for (int t = 0; t < T; ++t) {
        const size_t roff = base + (size_t)t * C;
        if      (rg == 0) sw[col] = (float)wbuf[roff + col];
        else if (rg == 1) sk[col] = (float)kbuf[roff + col];
        else if (rg == 2) sv[col] = (float)vbuf[roff + col];
        else              sr[col] = (float)rbuf[roff + col];
        __syncthreads();

        const float vj = sv[col];
        float part = 0.f;
        #pragma unroll
        for (int i = 0; i < 16; ++i) {
            const int s = rg * 16 + i;
            float x = st[i];
            x = fmaf(-sw[s], x, x);
            x = fmaf(sk[s], vj, x);
            st[i] = x;
            part = fmaf(sr[s], x, part);
        }
        sred[tid] = part;
        __syncthreads();
        if (rg == 0) {
            float y = sred[col] + sred[64 + col] + sred[128 + col] + sred[192 + col];
            ybuf[roff + col] = (_Float16)y;
        }
    }

    #pragma unroll
    for (int i = 0; i < 16; ++i) {
        const int s = rg * 16 + i;
        state_out[((size_t)bh * 64 + s) * 64 + col] = st[i];
    }
}

// ---------------------------------------------------------------------------
extern "C" void kernel_launch(void* const* d_in, const int* in_sizes, int n_in,
                              void* d_out, int out_size, void* d_ws, size_t ws_size,
                              hipStream_t stream)
{
    const float* x     = (const float*)d_in[0];
    const float* Wx    = (const float*)d_in[1];
    const float* Ww    = (const float*)d_in[2];
    const float* bw    = (const float*)d_in[3];
    const float* Wk    = (const float*)d_in[4];
    const float* Wv    = (const float*)d_in[5];
    const float* Wr    = (const float*)d_in[6];
    const float* Wo    = (const float*)d_in[7];
    const float* gamma = (const float*)d_in[8];
    const float* beta  = (const float*)d_in[9];
    float* out = (float*)d_out;

    const int C = DM, T = SEQ, H = HEADS;
    const size_t MT = (size_t)BATCH * T;

    char* p = (char*)d_ws;
    auto alloc_h = [&](size_t elems) {
        _Float16* q = (_Float16*)p;
        p += elems * sizeof(_Float16);
        return q;
    };
    _Float16* xn  = alloc_h(MT * C);
    _Float16* xx  = alloc_h(MT * C);
    _Float16* wb_ = alloc_h(MT * C);
    _Float16* kb_ = alloc_h(MT * C);
    _Float16* vb_ = alloc_h(MT * C);
    _Float16* rb_ = alloc_h(MT * C);
    _Float16* yb_ = alloc_h(MT * C);
    _Float16* Wxt = alloc_h((size_t)C * C);
    _Float16* Wwt = alloc_h((size_t)C * C);
    _Float16* Wkt = alloc_h((size_t)C * C);
    _Float16* Wvt = alloc_h((size_t)C * C);
    _Float16* Wrt = alloc_h((size_t)C * C);
    _Float16* Wot = alloc_h((size_t)C * C);

    const dim3 tb(256);

    const dim3 tg(C / 32, C / 32);
    transpose_to_f16<<<tg, tb, 0, stream>>>(Wx, Wxt, C);
    transpose_to_f16<<<tg, tb, 0, stream>>>(Ww, Wwt, C);
    transpose_to_f16<<<tg, tb, 0, stream>>>(Wk, Wkt, C);
    transpose_to_f16<<<tg, tb, 0, stream>>>(Wv, Wvt, C);
    transpose_to_f16<<<tg, tb, 0, stream>>>(Wr, Wrt, C);
    transpose_to_f16<<<tg, tb, 0, stream>>>(Wo, Wot, C);

    layernorm_to_f16<<<dim3((unsigned)MT), tb, 0, stream>>>(x, gamma, beta, xn, C);

    const dim3 gg(C / 64, (unsigned)(MT / 128));
    gemm_wmma_f16<EPI_F16><<<gg, tb, 0, stream>>>(xn, Wxt, nullptr, xx,  nullptr, (int)MT, C);
    gemm_wmma_f16<EPI_F16><<<gg, tb, 0, stream>>>(xn, Wkt, nullptr, kb_, nullptr, (int)MT, C);
    gemm_wmma_f16<EPI_F16><<<gg, tb, 0, stream>>>(xn, Wvt, nullptr, vb_, nullptr, (int)MT, C);
    gemm_wmma_f16<EPI_F16_SIG><<<gg, tb, 0, stream>>>(xn, Wrt, nullptr, rb_, nullptr, (int)MT, C);

    gemm_wmma_f16<EPI_F16_BIAS_SIG><<<gg, tb, 0, stream>>>(xx, Wwt, bw, wb_, nullptr, (int)MT, C);

    wkv_scan<<<dim3(BATCH * H), tb, 0, stream>>>(wb_, kb_, vb_, rb_, yb_,
                                                 out + MT * (size_t)C, T, C, H);

    gemm_wmma_f16<EPI_F32><<<gg, tb, 0, stream>>>(yb_, Wot, nullptr, nullptr, out, (int)MT, C);
}
